// Conductor_23158463660727
// MI455X (gfx1250) — matrix operand order repaired and checked by reading, not compiled, not run on hardware
//
#include <hip/hip_runtime.h>
#include <hip/hip_bf16.h>
#include <math.h>

typedef __attribute__((ext_vector_type(16))) _Float16 v16h;
typedef __attribute__((ext_vector_type(8)))  float    v8f;

#define B_   1024
#define D_   1024
#define H_   1024
#define G_   4096   // 4*H, gate width
#define HC_  2048   // concatenated [h0|h1] width
#define BAR_ 32

// ---- WMMA GEMM: C[m,n] = (Cin?Cin[m,n]:0) + bias[n] + sum_k A[m,k]*W[n,k] ----
// A: (M x K) f16, row stride lda. W: (N x K) f16, row stride K (multiply by W^T).
// C/Cin: f32 with row stride ldc. bias must be non-null (use a zero vector).
// M%128==0, N%128==0, K%32==0.
// Block: 256 threads = 8 wave32s; block tile 128x128; wave tile 32x64 (2x4 frags).
// Double-buffered LDS; next K-slab global loads issued before current compute.
#define BM 128
#define BN 128
#define BK 32
#define PAD 40   // halfs per padded LDS row (80B: 16B-aligned, bank-conflict-free)

__global__ __launch_bounds__(256)
void gemm_wmma_f16(float* C, const float* Cin,
                   const float* __restrict__ bias,
                   const _Float16* __restrict__ A,
                   const _Float16* __restrict__ W,
                   int M, int N, int K, int lda, int ldc)
{
    __shared__ __align__(16) _Float16 As[2][BM * PAD];
    __shared__ __align__(16) _Float16 Ws[2][BN * PAD];

    const int tid  = threadIdx.x;
    const int lane = tid & 31;
    const int wave = tid >> 5;
    const int wm   = wave & 3;    // 0..3 : 32-row slab
    const int wn   = wave >> 2;   // 0..1 : 64-col slab
    const int bm0  = blockIdx.x * BM;
    const int bn0  = blockIdx.y * BN;
    const int l15  = lane & 15;
    const int lhi  = lane >> 4;   // 0 or 1

    // Staging: each thread covers rows r and r+64 (A and W tiles), chunk cc (8 halfs)
    const int r  = tid >> 2;
    const int cc = tid & 3;
    const _Float16* Abase = A + (size_t)(bm0 + r) * lda + cc * 8;
    const _Float16* Wbase = W + (size_t)(bn0 + r) * K + cc * 8;
    const size_t strideRA = (size_t)64 * lda;
    const size_t strideRW = (size_t)64 * K;
    const int sidx  = r * PAD + cc * 8;
    const int sidx2 = (r + 64) * PAD + cc * 8;

    // ---- prologue: stage K-slab 0 into buffer 0 ----
    {
        uint4 a0 = *(const uint4*)(Abase);
        uint4 a1 = *(const uint4*)(Abase + strideRA);
        uint4 w0 = *(const uint4*)(Wbase);
        uint4 w1 = *(const uint4*)(Wbase + strideRW);
        *(uint4*)&As[0][sidx]  = a0;
        *(uint4*)&As[0][sidx2] = a1;
        *(uint4*)&Ws[0][sidx]  = w0;
        *(uint4*)&Ws[0][sidx2] = w1;
    }
    __syncthreads();

    v8f acc[2][4] = {};
    const int nk = K / BK;

    for (int k = 0; k < nk; ++k) {
        const int cur = k & 1;
        const int nxt = cur ^ 1;
        const bool pf = (k + 1 < nk);

        // Issue next slab's global loads early (independent registers -> they
        // pipeline, and the single wait lands after the WMMA block below).
        uint4 pa0 = {}, pa1 = {}, pw0 = {}, pw1 = {};
        if (pf) {
            const _Float16* Ap = Abase + (size_t)(k + 1) * BK;
            const _Float16* Wp = Wbase + (size_t)(k + 1) * BK;
            pa0 = *(const uint4*)(Ap);
            pa1 = *(const uint4*)(Ap + strideRA);
            pw0 = *(const uint4*)(Wp);
            pw1 = *(const uint4*)(Wp + strideRW);
        }

        // Fragment loads (ISA 7.12.2 layouts), then 8 WMMAs.
        v16h af[2], bf[4];
        #pragma unroll
        for (int fi = 0; fi < 2; ++fi) {
            // A frag 16x32: lane L<16 = row L, K {0..7,16..23}; lane L+16: K {8..15,24..31}
            int ra = wm * 32 + fi * 16 + l15;
            ((uint4*)&af[fi])[0] = *(const uint4*)(&As[cur][ra * PAD + lhi * 8]);
            ((uint4*)&af[fi])[1] = *(const uint4*)(&As[cur][ra * PAD + 16 + lhi * 8]);
        }
        #pragma unroll
        for (int fj = 0; fj < 4; ++fj) {
            // B frag 32x16: lane n<16 = col n holds K {0..7,16..23}; lane n+16 holds K {8..15,24..31}.
            // B[k][n] = W[n][k] -> contiguous run of W's row n.
            int rb = wn * 64 + fj * 16 + l15;
            ((uint4*)&bf[fj])[0] = *(const uint4*)(&Ws[cur][rb * PAD + lhi * 8]);
            ((uint4*)&bf[fj])[1] = *(const uint4*)(&Ws[cur][rb * PAD + 16 + lhi * 8]);
        }
        #pragma unroll
        for (int fi = 0; fi < 2; ++fi)
            #pragma unroll
            for (int fj = 0; fj < 4; ++fj)
                acc[fi][fj] = __builtin_amdgcn_wmma_f32_16x16x32_f16(
                    false, af[fi], false, bf[fj], (short)0, acc[fi][fj], false, false);

        if (pf) {
            *(uint4*)&As[nxt][sidx]  = pa0;
            *(uint4*)&As[nxt][sidx2] = pa1;
            *(uint4*)&Ws[nxt][sidx]  = pw0;
            *(uint4*)&Ws[nxt][sidx2] = pw1;
        }
        __syncthreads();
    }

    // ---- epilogue (ISA C/D layout: lane n<16 -> col n, VGPR r -> row r, +8 hi half)
    // One uniform branch on Cin; bias preloaded; stores share a row pointer with
    // immediate offsets (fj*64B).
    const int colbase = bn0 + wn * 64 + l15;
    float bv[4];
    #pragma unroll
    for (int fj = 0; fj < 4; ++fj) bv[fj] = bias[colbase + fj * 16];

    if (Cin) {
        #pragma unroll
        for (int fi = 0; fi < 2; ++fi)
            #pragma unroll
            for (int rr = 0; rr < 8; ++rr) {
                int row = bm0 + wm * 32 + fi * 16 + lhi * 8 + rr;
                float*       cp = C   + (size_t)row * ldc + colbase;
                const float* ip = Cin + (size_t)row * ldc + colbase;
                #pragma unroll
                for (int fj = 0; fj < 4; ++fj)
                    cp[fj * 16] = acc[fi][fj][rr] + bv[fj] + ip[fj * 16];
            }
    } else {
        #pragma unroll
        for (int fi = 0; fi < 2; ++fi)
            #pragma unroll
            for (int rr = 0; rr < 8; ++rr) {
                int row = bm0 + wm * 32 + fi * 16 + lhi * 8 + rr;
                float* cp = C + (size_t)row * ldc + colbase;
                #pragma unroll
                for (int fj = 0; fj < 4; ++fj)
                    cp[fj * 16] = acc[fi][fj][rr] + bv[fj];
            }
    }
}

// ---- BatchNorm (training stats) + init hcat=[zn|zn] (f16) and c0,c1 (f32) ----
__global__ __launch_bounds__(256)
void batchnorm_init(const float* __restrict__ z,
                    const float* __restrict__ gamma,
                    const float* __restrict__ beta,
                    _Float16* __restrict__ hcat,
                    float* __restrict__ c0, float* __restrict__ c1)
{
    __shared__ float ssum[256], ssq[256];
    const int d = blockIdx.x, tid = threadIdx.x;
    float s = 0.f, q = 0.f;
    for (int b = tid; b < B_; b += 256) {
        float v = z[(size_t)b * D_ + d];
        s += v; q += v * v;
    }
    ssum[tid] = s; ssq[tid] = q; __syncthreads();
    for (int off = 128; off > 0; off >>= 1) {
        if (tid < off) { ssum[tid] += ssum[tid + off]; ssq[tid] += ssq[tid + off]; }
        __syncthreads();
    }
    float mean = ssum[0] * (1.0f / B_);
    float var  = ssq[0] * (1.0f / B_) - mean * mean;
    float scale = gamma[d] * rsqrtf(var + 1e-5f);
    float shift = beta[d] - mean * scale;
    for (int b = tid; b < B_; b += 256) {
        float v = z[(size_t)b * D_ + d] * scale + shift;
        hcat[(size_t)b * HC_ + d]      = (_Float16)v;   // h0 slice
        hcat[(size_t)b * HC_ + H_ + d] = (_Float16)v;   // h1 slice
        c0[(size_t)b * H_ + d] = v;
        c1[(size_t)b * H_ + d] = v;
    }
}

// ---- LSTM pointwise cell: gate order i,f,g,o; h written into strided slice ----
__device__ __forceinline__ float sigm(float x) { return 1.0f / (1.0f + __expf(-x)); }

__global__ __launch_bounds__(256)
void lstm_cell(const float* __restrict__ gates, float* __restrict__ c,
               _Float16* __restrict__ h_out, int hstride)
{
    int idx = blockIdx.x * 256 + threadIdx.x;      // 0 .. B*H-1
    int b = idx >> 10, n = idx & (H_ - 1);
    const float* g = gates + (size_t)b * G_;
    float ig = sigm(g[n]);
    float fg = sigm(g[n + H_]);
    float gg = tanhf(g[n + 2 * H_]);
    float og = sigm(g[n + 3 * H_]);
    float cn = fg * c[idx] + ig * gg;
    c[idx] = cn;
    h_out[(size_t)b * hstride + n] = (_Float16)(og * tanhf(cn));
}

// ---- small helpers ----
__global__ __launch_bounds__(256)
void f32_to_f16(const float* __restrict__ in, _Float16* __restrict__ out, int n)
{
    int i = blockIdx.x * 256 + threadIdx.x;
    if (i < n) out[i] = (_Float16)in[i];
}

// out (G_ x 2H): row n = [wa row n | wb row n], converted to f16
__global__ __launch_bounds__(256)
void concat_w_f16(const float* __restrict__ wa, const float* __restrict__ wb,
                  _Float16* __restrict__ out)
{
    int i = blockIdx.x * 256 + threadIdx.x;        // over G_*2H
    int n = i >> 11, k = i & (HC_ - 1);
    float v = (k < H_) ? wa[(size_t)n * H_ + k] : wb[(size_t)n * H_ + (k - H_)];
    out[i] = (_Float16)v;
}

__global__ __launch_bounds__(256)
void vec_add(const float* __restrict__ a, const float* __restrict__ b,
             float* __restrict__ out, int n)
{
    int i = blockIdx.x * 256 + threadIdx.x;
    if (i < n) out[i] = a[i] + b[i];
}

__global__ __launch_bounds__(256)
void vec_zero(float* __restrict__ out, int n)
{
    int i = blockIdx.x * 256 + threadIdx.x;
    if (i < n) out[i] = 0.0f;
}

extern "C" void kernel_launch(void* const* d_in, const int* in_sizes, int n_in,
                              void* d_out, int out_size, void* d_ws, size_t ws_size,
                              hipStream_t stream)
{
    const float* z      = (const float*)d_in[0];
    const float* gamma  = (const float*)d_in[1];
    const float* beta   = (const float*)d_in[2];
    const float* W_ih0  = (const float*)d_in[3];
    const float* W_hh0  = (const float*)d_in[4];
    const float* b_ih0  = (const float*)d_in[5];
    const float* b_hh0  = (const float*)d_in[6];
    const float* W_ih1  = (const float*)d_in[7];
    const float* W_hh1  = (const float*)d_in[8];
    const float* b_ih1  = (const float*)d_in[9];
    const float* b_hh1  = (const float*)d_in[10];
    const float* W_lin  = (const float*)d_in[11];
    const float* b_lin  = (const float*)d_in[12];
    float* out = (float*)d_out;

    // ---- workspace carve-up (256B aligned) ----
    size_t off = 0;
    auto carve = [&](size_t bytes) -> char* {
        char* p = (char*)d_ws + off;
        off += (bytes + 255) & ~(size_t)255;
        return p;
    };
    _Float16* wih0h = (_Float16*)carve((size_t)G_ * D_ * 2);   // W_ih0 f16
    _Float16* whh0h = (_Float16*)carve((size_t)G_ * H_ * 2);   // W_hh0 f16
    _Float16* wcat1 = (_Float16*)carve((size_t)G_ * HC_ * 2);  // [W_ih1|W_hh1] f16
    _Float16* wlinh = (_Float16*)carve((size_t)D_ * H_ * 2);   // W_lin f16
    float* b0s      = (float*)carve((size_t)G_ * 4);
    float* b1s      = (float*)carve((size_t)G_ * 4);
    float* zbias    = (float*)carve((size_t)G_ * 4);
    _Float16* hcat  = (_Float16*)carve((size_t)B_ * HC_ * 2);  // [h0|h1] f16
    float* c0       = (float*)carve((size_t)B_ * H_ * 4);
    float* c1       = (float*)carve((size_t)B_ * H_ * 4);
    float* x0proj   = (float*)carve((size_t)B_ * G_ * 4);
    float* gates0   = (float*)carve((size_t)B_ * G_ * 4);
    float* gates1   = (float*)carve((size_t)B_ * G_ * 4);
    (void)ws_size; (void)n_in; (void)in_sizes; (void)out_size;

    // ---- setup: weight conversion/concat, bias folding, batchnorm ----
    const int WN = G_ * D_;   // 4M elements
    f32_to_f16<<<(WN + 255) / 256, 256, 0, stream>>>(W_ih0, wih0h, WN);
    f32_to_f16<<<(WN + 255) / 256, 256, 0, stream>>>(W_hh0, whh0h, WN);
    concat_w_f16<<<(G_ * HC_) / 256, 256, 0, stream>>>(W_ih1, W_hh1, wcat1);
    f32_to_f16<<<(D_ * H_ + 255) / 256, 256, 0, stream>>>(W_lin, wlinh, D_ * H_);
    vec_add<<<(G_ + 255) / 256, 256, 0, stream>>>(b_ih0, b_hh0, b0s, G_);
    vec_add<<<(G_ + 255) / 256, 256, 0, stream>>>(b_ih1, b_hh1, b1s, G_);
    vec_zero<<<(G_ + 255) / 256, 256, 0, stream>>>(zbias, G_);
    batchnorm_init<<<D_, 256, 0, stream>>>(z, gamma, beta, hcat, c0, c1);

    // x0proj = zn @ W_ih0^T + (b_ih0 + b_hh0)   (zn lives in hcat's h0 slice)
    {
        dim3 grid(B_ / BM, G_ / BN);
        gemm_wmma_f16<<<grid, 256, 0, stream>>>(
            x0proj, nullptr, b0s, hcat, wih0h, B_, G_, D_, HC_, G_);
    }

    const dim3 gridGate(B_ / BM, G_ / BN);   // 8 x 32
    const dim3 gridOut(B_ / BM, D_ / BN);    // 8 x 8
    const int cellBlocks = (B_ * H_) / 256;  // 4096

    for (int t = 0; t < BAR_; ++t) {
        // layer 0: gates0 = x0proj + h0 @ W_hh0^T
        gemm_wmma_f16<<<gridGate, 256, 0, stream>>>(
            gates0, x0proj, zbias, hcat, whh0h, B_, G_, H_, HC_, G_);
        lstm_cell<<<cellBlocks, 256, 0, stream>>>(gates0, c0, hcat, HC_);

        // layer 1 (fused): gates1 = [h0|h1] @ [W_ih1|W_hh1]^T + (b_ih1+b_hh1)
        gemm_wmma_f16<<<gridGate, 256, 0, stream>>>(
            gates1, nullptr, b1s, hcat, wcat1, B_, G_, HC_, HC_, G_);
        lstm_cell<<<cellBlocks, 256, 0, stream>>>(gates1, c1, hcat + H_, HC_);

        // out[:, t, :] = h1 @ W_lin^T + b_lin  (row stride BAR*D, offset t*D)
        gemm_wmma_f16<<<gridOut, 256, 0, stream>>>(
            out + (size_t)t * D_, nullptr, b_lin, hcat + H_, wlinh,
            B_, D_, H_, HC_, BAR_ * D_);
    }
}
